// TransformerEncoderBlock_70944269795917
// MI455X (gfx1250) — compile-verified, matrix operand order checked by reading, not run on hardware
//
#include <hip/hip_runtime.h>
#include <hip/hip_bf16.h>
#include <math.h>

// ---------------- types ----------------
typedef __bf16 bf16_t;
typedef __attribute__((ext_vector_type(16))) __bf16 v16bf;
typedef __attribute__((ext_vector_type(8)))  float  v8f;
typedef __attribute__((ext_vector_type(4)))  unsigned int u32x4;
typedef __attribute__((ext_vector_type(4)))  int i32x4;
typedef __attribute__((ext_vector_type(8)))  int i32x8;

union AFrag { v16bf v; uint4 u[2]; };

__device__ __forceinline__ bf16_t f2bf(float f) { return (bf16_t)f; }

__device__ __forceinline__ v8f wmma_bf16(v16bf a, v16bf b, v8f c) {
  // (neg_a, A, neg_b, B, c_mod, C, reuse_a, reuse_b)
  return __builtin_amdgcn_wmma_f32_16x16x32_bf16(false, a, false, b, (short)0, c, false, false);
}

// ---------------- problem constants ----------------
#define BB 8
#define SS 1024
#define EE 1024
#define HH 16
#define DD 64
#define FF2 4096

// ---------------- TDM: 2D tile global -> LDS ----------------
#if __has_builtin(__builtin_amdgcn_tensor_load_to_lds)
#define USE_TDM 1
#else
#define USE_TDM 0
#endif

#if USE_TDM
// Build D# for a 2D bf16 tile: tile_d0 contiguous elements per row,
// tile_d1 rows, row stride = stride0 elements, packed per ISA 8.3/8.4.
__device__ __forceinline__ void tdm_load_2d(unsigned lds_addr, const void* gptr,
                                            unsigned tile_d0, unsigned tile_d1,
                                            unsigned tensor_d0, unsigned tensor_d1,
                                            unsigned long long stride0) {
  unsigned long long ga = (unsigned long long)(uintptr_t)gptr;
  u32x4 g0;
  g0[0] = 1u;                                          // count=1, user mode
  g0[1] = lds_addr;                                    // LDS byte address
  g0[2] = (unsigned)(ga & 0xffffffffull);              // global_addr[31:0]
  g0[3] = (unsigned)((ga >> 32) & 0x1ffffffull) | (2u << 30); // addr[56:32], type=2
  i32x8 g1;
  g1[0] = (int)(1u << 16);                             // data_size=1 (2 bytes)
  g1[1] = (int)((tensor_d0 & 0xffffu) << 16);          // dim0[15:0] @ bits63:48
  g1[2] = (int)(((tensor_d0 >> 16) & 0xffffu) | ((tensor_d1 & 0xffffu) << 16));
  g1[3] = (int)(((tensor_d1 >> 16) & 0xffffu) | (tile_d0 << 16));
  g1[4] = (int)tile_d1;                                // tile_dim1 (tile_dim2=0)
  g1[5] = (int)(stride0 & 0xffffffffull);              // dim0_stride[31:0]
  g1[6] = (int)((stride0 >> 32) & 0xffffull);          // dim0_stride[47:32]
  g1[7] = 0;                                           // dim1_stride (unused, 2D)
  i32x4 z4 = {0, 0, 0, 0};
#if __has_include(<hip/amd_detail/amd_gfx1250_TDM.h>)
  i32x8 z8 = {0, 0, 0, 0, 0, 0, 0, 0};
  __builtin_amdgcn_tensor_load_to_lds(g0, g1, z4, z4, z8, 0);   // clang-23 form
#else
  __builtin_amdgcn_tensor_load_to_lds(g0, g1, z4, z4, 0);       // ROCm 7.2 form
#endif
}
#endif

// =====================================================================
// Weight packing / conversion kernels (weights stored TRANSPOSED [N,K])
// =====================================================================
__global__ __launch_bounds__(256) void pack_qkv_w(const float* __restrict__ Wq,
                                                  const float* __restrict__ Wk,
                                                  const float* __restrict__ Wv,
                                                  bf16_t* __restrict__ out) {
  int idx = blockIdx.x * 256 + threadIdx.x;          // over 3072*1024, out[n][e]
  int n = idx >> 10;
  int e = idx & 1023;
  int s = n >> 10;                                   // 0=q 1=k 2=v
  int c = n & 1023;
  int h = c >> 6, d = c & 63;
  const float* W = (s == 0) ? Wq : ((s == 1) ? Wk : Wv);
  out[idx] = f2bf(W[((size_t)h * EE + e) * DD + d]); // [H,E,D] -> [3*H*D, E]
}

__global__ __launch_bounds__(256) void pack_qkv_bias(const float* __restrict__ bq,
                                                     const float* __restrict__ bk,
                                                     const float* __restrict__ bv,
                                                     float* __restrict__ out) {
  int i = blockIdx.x * 256 + threadIdx.x;
  if (i >= 3072) return;
  int s = i >> 10, c = i & 1023;
  const float* b = (s == 0) ? bq : ((s == 1) ? bk : bv);
  out[i] = b[c];
}

// out[n*K + k] = in[k*N + n]  (fp32 -> bf16, transpose)
__global__ __launch_bounds__(256) void cvt_t_bf16(const float* __restrict__ in,
                                                  bf16_t* __restrict__ out,
                                                  int K, int N) {
  int idx = blockIdx.x * 256 + threadIdx.x;
  int n = idx / K, k = idx - n * K;
  out[idx] = f2bf(in[(size_t)k * N + n]);
}

// Transpose V heads: qkv[B*S, 3072] (v section) -> vt[B,H,D,S] bf16
__global__ __launch_bounds__(256) void transpose_v(const bf16_t* __restrict__ qkv,
                                                   bf16_t* __restrict__ vt) {
  int idx = blockIdx.x * 256 + threadIdx.x;          // over B*H*D*S = 2^23
  int s = idx & 1023;
  int d = (idx >> 10) & 63;
  int h = (idx >> 16) & 15;
  int b = idx >> 20;
  vt[idx] = qkv[((size_t)(b * SS + s)) * 3072 + 2048 + h * 64 + d];
}

// =====================================================================
// LayerNorm: fp32 [rows,1024] -> bf16, one block per row
// =====================================================================
__global__ __launch_bounds__(256) void layernorm_bf16(const float* __restrict__ x,
                                                      const float* __restrict__ g,
                                                      const float* __restrict__ be,
                                                      bf16_t* __restrict__ out) {
  __shared__ float rs[256], rq[256];
  const int tid = threadIdx.x;
  const float* xr = x + (size_t)blockIdx.x * EE;
  float vals[4], s = 0.f, s2 = 0.f;
#pragma unroll
  for (int i = 0; i < 4; i++) {
    float v = xr[tid + i * 256];
    vals[i] = v; s += v; s2 += v * v;
  }
  rs[tid] = s; rq[tid] = s2;
  __syncthreads();
  for (int off = 128; off > 0; off >>= 1) {
    if (tid < off) { rs[tid] += rs[tid + off]; rq[tid] += rq[tid + off]; }
    __syncthreads();
  }
  float mu   = rs[0] * (1.0f / EE);
  float var  = rq[0] * (1.0f / EE) - mu * mu;
  float rinv = rsqrtf(var + 1e-5f);
  bf16_t* orow = out + (size_t)blockIdx.x * EE;
#pragma unroll
  for (int i = 0; i < 4; i++) {
    int c = tid + i * 256;
    orow[c] = f2bf((vals[i] - mu) * rinv * g[c] + be[c]);
  }
}

// =====================================================================
// Tiled bf16 WMMA GEMM: C[M,N] = A[M,K] x Bt[N,K]^T (+bias, +res, gelu)
// Block 256 thr = 8 waves; block tile 128x128; wave tile 64x32; K-step 32.
// Tiles staged global->LDS by the Tensor Data Mover (wave 0), or by
// per-thread b128 copies when the TDM builtin is unavailable.
// =====================================================================
template <bool GELU_, bool RES, bool OUTBF>
__global__ __launch_bounds__(256) void gemm_bf16(const bf16_t* __restrict__ A,
                                                 const bf16_t* __restrict__ Bt,
                                                 const float* __restrict__ bias,
                                                 const float* __restrict__ res,
                                                 float* __restrict__ outf,
                                                 bf16_t* __restrict__ outbf,
                                                 int M, int N, int K) {
  __shared__ __align__(16) bf16_t As[128 * 32];      // [row][k]
  __shared__ __align__(16) bf16_t Bs[128 * 32];      // [n][k]
  const int tid  = threadIdx.x;
  const int lane = tid & 31, wid = tid >> 5;
  const int wm = wid >> 2, wn = wid & 3;             // 2 x 4 wave grid
  const int ln = lane & 15;
  const int kb = (lane < 16) ? 0 : 8;
  const int rowBase = blockIdx.y * 128, colBase = blockIdx.x * 128;

  v8f acc[4][2] = {};

#if USE_TDM
  const unsigned ldsA = (unsigned)(uintptr_t)(void*)As;
  const unsigned ldsB = (unsigned)(uintptr_t)(void*)Bs;
#else
  const int lrow = tid >> 1, lk0 = (tid & 1) * 16;   // tile loader mapping
#endif

  for (int kt = 0; kt < K; kt += 32) {
#if USE_TDM
    __syncthreads();                                 // LDS free to overwrite
    if (wid == 0) {
      tdm_load_2d(ldsA, A  + (size_t)rowBase * K + kt, 32, 128,
                  (unsigned)K, (unsigned)M, (unsigned long long)K);
      tdm_load_2d(ldsB, Bt + (size_t)colBase * K + kt, 32, 128,
                  (unsigned)K, (unsigned)N, (unsigned long long)K);
      __builtin_amdgcn_s_wait_tensorcnt(0);
    }
    __syncthreads();                                 // tiles visible
#else
    const bf16_t* ap = A  + (size_t)(rowBase + lrow) * K + kt + lk0;
    uint4 a0 = *(const uint4*)(ap);
    uint4 a1 = *(const uint4*)(ap + 8);
    const bf16_t* bp = Bt + (size_t)(colBase + lrow) * K + kt + lk0;
    uint4 b0 = *(const uint4*)(bp);
    uint4 b1 = *(const uint4*)(bp + 8);
    __syncthreads();
    *(uint4*)(As + lrow * 32 + lk0)     = a0;
    *(uint4*)(As + lrow * 32 + lk0 + 8) = a1;
    *(uint4*)(Bs + lrow * 32 + lk0)     = b0;
    *(uint4*)(Bs + lrow * 32 + lk0 + 8) = b1;
    __syncthreads();
#endif

    AFrag af[4], bfrag[2];
#pragma unroll
    for (int mt = 0; mt < 4; mt++) {
      const bf16_t* p = As + (wm * 64 + mt * 16 + ln) * 32;
      af[mt].u[0] = *(const uint4*)(p + kb);
      af[mt].u[1] = *(const uint4*)(p + kb + 16);
    }
#pragma unroll
    for (int nt = 0; nt < 2; nt++) {
      const bf16_t* p = Bs + (wn * 32 + nt * 16 + ln) * 32;
      bfrag[nt].u[0] = *(const uint4*)(p + kb);
      bfrag[nt].u[1] = *(const uint4*)(p + kb + 16);
    }
#pragma unroll
    for (int mt = 0; mt < 4; mt++)
#pragma unroll
      for (int nt = 0; nt < 2; nt++)
        acc[mt][nt] = wmma_bf16(af[mt].v, bfrag[nt].v, acc[mt][nt]);
  }

  const int rOff = (lane < 16) ? 0 : 8;
#pragma unroll
  for (int mt = 0; mt < 4; mt++)
#pragma unroll
    for (int nt = 0; nt < 2; nt++)
#pragma unroll
      for (int r = 0; r < 8; r++) {
        int row = rowBase + wm * 64 + mt * 16 + r + rOff;
        int col = colBase + wn * 32 + nt * 16 + ln;
        float v = acc[mt][nt][r] + bias[col];
        if constexpr (RES)   v += res[(size_t)row * N + col];
        if constexpr (GELU_) v = 0.5f * v * (1.0f + erff(v * 0.70710678118654752f));
        if constexpr (OUTBF) outbf[(size_t)row * N + col] = f2bf(v);
        else                 outf[(size_t)row * N + col]  = v;
      }
}

// =====================================================================
// Flash attention, register-resident. 1 wave = 1 (b,h,16-query tile).
// Scores computed TRANSPOSED (S^T = K x Q^T) so that P^T in C-layout is
// bit-for-bit the B-fragment of ctx^T = V^T x P^T (no shuffles needed).
// =====================================================================
__global__ __launch_bounds__(128) void attn_kernel(const bf16_t* __restrict__ qkv,
                                                   const bf16_t* __restrict__ vt,
                                                   bf16_t* __restrict__ ctx) {
  const int lane = threadIdx.x & 31;
  const int wid  = threadIdx.x >> 5;
  const int tile = blockIdx.x * 4 + wid;             // B*H*(S/16) tiles
  const int qt = tile & 63;
  const int h  = (tile >> 6) & 15;
  const int b  = tile >> 10;
  const int ln = lane & 15;
  const int kb = (lane < 16) ? 0 : 8;

  // Q as B-fragment for S^T GEMM: lane->query col, elements->d
  AFrag qf[2];
  const bf16_t* qbase = qkv + ((size_t)(b * SS + qt * 16 + ln)) * 3072 + h * 64;
#pragma unroll
  for (int c = 0; c < 2; c++) {
    qf[c].u[0] = *(const uint4*)(qbase + c * 32 + kb);
    qf[c].u[1] = *(const uint4*)(qbase + c * 32 + kb + 16);
  }

  v8f o[4] = {};
  float m_run = -3.0e38f, l_run = 0.0f;
  const bf16_t* kbp = qkv + (size_t)b * SS * 3072 + 1024 + h * 64;
  const bf16_t* vtb = vt + ((size_t)((b * HH + h) * DD)) * SS;

  for (int kt = 0; kt < 64; kt += 2) {               // 32 keys per step
    // K rows as A-fragments (lane->key row, elements->d)
    AFrag ka[2][2];
#pragma unroll
    for (int s2 = 0; s2 < 2; s2++) {
      const bf16_t* kp = kbp + (size_t)((kt + s2) * 16 + ln) * 3072;
#pragma unroll
      for (int c = 0; c < 2; c++) {
        ka[s2][c].u[0] = *(const uint4*)(kp + c * 32 + kb);
        ka[s2][c].u[1] = *(const uint4*)(kp + c * 32 + kb + 16);
      }
    }
    v8f s0 = {}, s1 = {};
    s0 = wmma_bf16(ka[0][0].v, qf[0].v, s0);
    s0 = wmma_bf16(ka[0][1].v, qf[1].v, s0);
    s1 = wmma_bf16(ka[1][0].v, qf[0].v, s1);
    s1 = wmma_bf16(ka[1][1].v, qf[1].v, s1);

    // online softmax (stats are per query == per lane; one xor-16 combine)
    float mx = -3.0e38f;
#pragma unroll
    for (int r = 0; r < 8; r++) {
      s0[r] *= 0.125f; s1[r] *= 0.125f;              // 1/sqrt(64)
      mx = fmaxf(mx, fmaxf(s0[r], s1[r]));
    }
    mx = fmaxf(mx, __shfl_xor(mx, 16, 32));
    float mnew = fmaxf(m_run, mx);
    float f = __expf(m_run - mnew);
    float p0[8], p1[8], rsum = 0.f;
#pragma unroll
    for (int r = 0; r < 8; r++) {
      p0[r] = __expf(s0[r] - mnew);
      p1[r] = __expf(s1[r] - mnew);
      rsum += p0[r] + p1[r];
    }
    rsum += __shfl_xor(rsum, 16, 32);
    l_run = l_run * f + rsum;
    m_run = mnew;
#pragma unroll
    for (int mt = 0; mt < 4; mt++)
#pragma unroll
      for (int r = 0; r < 8; r++) o[mt][r] *= f;

    // P^T C-layout == B-fragment layout: elems 0..7 <- tile0, 8..15 <- tile1
    AFrag pf;
#pragma unroll
    for (int r = 0; r < 8; r++) {
      pf.v[r]     = f2bf(p0[r]);
      pf.v[r + 8] = f2bf(p1[r]);
    }
    // ctx^T += V^T x P^T  (V^T A-fragments are contiguous loads from vt)
#pragma unroll
    for (int mt = 0; mt < 4; mt++) {
      const bf16_t* vp = vtb + (size_t)(mt * 16 + ln) * SS + kt * 16;
      AFrag va;
      va.u[0] = *(const uint4*)(vp + kb);
      va.u[1] = *(const uint4*)(vp + kb + 16);
      o[mt] = wmma_bf16(va.v, pf.v, o[mt]);
    }
  }

  float ro = 1.0f / l_run;
  const int rOff = (lane < 16) ? 0 : 8;
  bf16_t* op = ctx + ((size_t)(b * SS + qt * 16 + ln)) * EE + h * 64;
#pragma unroll
  for (int mt = 0; mt < 4; mt++)
#pragma unroll
    for (int r = 0; r < 8; r++)
      op[mt * 16 + r + rOff] = f2bf(o[mt][r] * ro);
}

// =====================================================================
// Host launcher
// =====================================================================
extern "C" void kernel_launch(void* const* d_in, const int* in_sizes, int n_in,
                              void* d_out, int out_size, void* d_ws, size_t ws_size,
                              hipStream_t stream) {
  const float* x    = (const float*)d_in[0];
  const float* Wq   = (const float*)d_in[1];
  const float* bq   = (const float*)d_in[2];
  const float* Wk   = (const float*)d_in[3];
  const float* bk   = (const float*)d_in[4];
  const float* Wv   = (const float*)d_in[5];
  const float* bv   = (const float*)d_in[6];
  const float* Wo   = (const float*)d_in[7];
  const float* bo   = (const float*)d_in[8];
  const float* g1   = (const float*)d_in[9];
  const float* be1  = (const float*)d_in[10];
  const float* g2   = (const float*)d_in[11];
  const float* be2  = (const float*)d_in[12];
  const float* W1   = (const float*)d_in[13];
  const float* c1   = (const float*)d_in[14];
  const float* W2   = (const float*)d_in[15];
  const float* c2   = (const float*)d_in[16];

  const size_t MROWS = (size_t)BB * SS;              // 8192
  char* p = (char*)d_ws;
  auto alloc = [&](size_t bytes) {
    void* r = (void*)p;
    p += (bytes + 255) & ~(size_t)255;
    return r;
  };
  bf16_t* xnB    = (bf16_t*)alloc(MROWS * EE * 2);
  bf16_t* WqkvT  = (bf16_t*)alloc((size_t)3072 * EE * 2);   // [N,K]
  float*  qkvb   = (float*) alloc(3072 * 4);
  bf16_t* WoT    = (bf16_t*)alloc((size_t)EE * EE * 2);     // [N,K]
  bf16_t* W1T    = (bf16_t*)alloc((size_t)FF2 * EE * 2);    // [N,K]
  bf16_t* W2T    = (bf16_t*)alloc((size_t)EE * FF2 * 2);    // [N,K]
  bf16_t* qkvB   = (bf16_t*)alloc(MROWS * 3072 * 2);
  bf16_t* vtB    = (bf16_t*)alloc((size_t)BB * HH * DD * SS * 2);
  bf16_t* ctxB   = (bf16_t*)alloc(MROWS * EE * 2);
  float*  out1   = (float*) alloc(MROWS * EE * 4);
  bf16_t* xn2B   = (bf16_t*)alloc(MROWS * EE * 2);
  bf16_t* hB     = (bf16_t*)alloc(MROWS * FF2 * 2);

  // 1) weight packing / conversion (transposed, deterministic every call)
  pack_qkv_w   <<<(3072 * EE) / 256, 256, 0, stream>>>(Wq, Wk, Wv, WqkvT);
  pack_qkv_bias<<<12, 256, 0, stream>>>(bq, bk, bv, qkvb);
  cvt_t_bf16   <<<(EE * EE) / 256, 256, 0, stream>>>(Wo, WoT, EE, EE);
  cvt_t_bf16   <<<(EE * FF2) / 256, 256, 0, stream>>>(W1, W1T, EE, FF2);
  cvt_t_bf16   <<<(FF2 * EE) / 256, 256, 0, stream>>>(W2, W2T, FF2, EE);

  // 2) LN1 -> xn (bf16)
  layernorm_bf16<<<(int)MROWS, 256, 0, stream>>>(x, g1, be1, xnB);

  // 3) fused QKV GEMM: [8192,1024] x [1024,3072] -> bf16
  gemm_bf16<false, false, true><<<dim3(3072 / 128, MROWS / 128), 256, 0, stream>>>(
      xnB, WqkvT, qkvb, nullptr, nullptr, qkvB, (int)MROWS, 3072, EE);

  // 4) transpose V heads for contiguous V^T fragments
  transpose_v<<<(BB * HH * DD * SS) / 256, 256, 0, stream>>>(qkvB, vtB);

  // 5) flash attention: B*H*(S/16)=8192 wave-tiles, 4 waves/block
  attn_kernel<<<(BB * HH * (SS / 16)) / 4, 128, 0, stream>>>(qkvB, vtB, ctxB);

  // 6) Wo GEMM + bias + residual(x) -> out1 (fp32)
  gemm_bf16<false, true, false><<<dim3(EE / 128, MROWS / 128), 256, 0, stream>>>(
      ctxB, WoT, bo, x, out1, nullptr, (int)MROWS, EE, EE);

  // 7) LN2
  layernorm_bf16<<<(int)MROWS, 256, 0, stream>>>(out1, g2, be2, xn2B);

  // 8) MLP up: GEMM + bias + exact GELU -> bf16
  gemm_bf16<true, false, true><<<dim3(FF2 / 128, MROWS / 128), 256, 0, stream>>>(
      xn2B, W1T, c1, nullptr, nullptr, hB, (int)MROWS, FF2, EE);

  // 9) MLP down: GEMM + bias + residual(out1) -> d_out (fp32)
  gemm_bf16<false, true, false><<<dim3(EE / 128, MROWS / 128), 256, 0, stream>>>(
      hB, W2T, c2, out1, (float*)d_out, nullptr, (int)MROWS, EE, FF2);

  (void)in_sizes; (void)n_in; (void)out_size; (void)ws_size;
}
